// BoxIMFDGCNN_31301721653644
// MI455X (gfx1250) — matrix operand order repaired and checked by reading, compiled, not run on hardware
//
#include <hip/hip_runtime.h>
#include <hip/hip_bf16.h>

typedef __attribute__((ext_vector_type(16))) _Float16 v16h;
typedef __attribute__((ext_vector_type(8)))  _Float16 v8h;
typedef __attribute__((ext_vector_type(8)))  float    v8f;

#define NN    16384
#define KNNK  10
#define EMBD  64
#define HIDD  128
#define DGCD  64
#define NCLSD 16

__device__ __forceinline__ v16h make_v16h(v8h lo, v8h hi) {
  return __builtin_shufflevector(lo, hi, 0,1,2,3,4,5,6,7,8,9,10,11,12,13,14,15);
}

// ---------------------------------------------------------------------------
// Prep: transpose EdgeConv weights to f16 [out_col][fan_in] for contiguous
// WMMA B-fragment loads.
// ---------------------------------------------------------------------------
__global__ void prep_weights_kernel(const float* __restrict__ We1,
                                    const float* __restrict__ We2,
                                    _Float16* __restrict__ WeT1,
                                    _Float16* __restrict__ WeT2) {
  int t = blockIdx.x * blockDim.x + threadIdx.x;
  if (t < 64 * 256) {
    int col = t / 256, k = t - col * 256;
    WeT1[t] = (_Float16)We1[k * 64 + col];          // We1: [256][64]
  } else if (t < 64 * 256 + 64 * 128) {
    int u = t - 64 * 256;
    int col = u / 128, k = u - col * 128;
    WeT2[u] = (_Float16)We2[k * 64 + col];          // We2: [128][64]
  }
}

// ---------------------------------------------------------------------------
// Embed (3x Linear+ReLU) + fusion (Linear+LeakyReLU). Emits h in f32 and f16
// plus per-row squared norm for the KNN distance trick.
// ---------------------------------------------------------------------------
__global__ __launch_bounds__(192)
void embed_fuse_kernel(const float* __restrict__ nf, const float* __restrict__ rf,
                       const float* __restrict__ tf,
                       const float* __restrict__ Wb, const float* __restrict__ bb,
                       const float* __restrict__ Wr, const float* __restrict__ br,
                       const float* __restrict__ Wt, const float* __restrict__ bt,
                       const float* __restrict__ Wf, const float* __restrict__ bf,
                       float* __restrict__ Hf, _Float16* __restrict__ Hh,
                       float* __restrict__ Hsq) {
  __shared__ float inbuf[136];   // 8 + 64 + 64
  __shared__ float emb[192];
  __shared__ float sqs;
  const int t = threadIdx.x;
  for (int n = blockIdx.x; n < NN; n += gridDim.x) {
    if (t < 8)        inbuf[t] = nf[n * 8 + t];
    else if (t < 72)  inbuf[t] = rf[n * 64 + (t - 8)];
    else if (t < 136) inbuf[t] = tf[n * 64 + (t - 72)];
    if (t == 0) sqs = 0.f;
    __syncthreads();

    float v;
    if (t < 64) {
      v = bb[t];
      #pragma unroll
      for (int i = 0; i < 8; ++i) v = fmaf(inbuf[i], Wb[i * 64 + t], v);
    } else if (t < 128) {
      const int o = t - 64;
      v = br[o];
      for (int i = 0; i < 64; ++i) v = fmaf(inbuf[8 + i], Wr[i * 64 + o], v);
    } else {
      const int o = t - 128;
      v = bt[o];
      for (int i = 0; i < 64; ++i) v = fmaf(inbuf[72 + i], Wt[i * 64 + o], v);
    }
    emb[t] = fmaxf(v, 0.f);
    __syncthreads();

    if (t < 128) {
      float h = bf[t];
      for (int ei = 0; ei < 192; ++ei) h = fmaf(emb[ei], Wf[ei * 128 + t], h);
      h = h > 0.f ? h : 0.01f * h;                 // leaky_relu, slope 0.01
      Hf[n * HIDD + t] = h;
      Hh[n * HIDD + t] = (_Float16)h;
      atomicAdd(&sqs, h * h);
    }
    __syncthreads();
    if (t == 0) Hsq[n] = sqs;
    __syncthreads();
  }
}

// ---------------------------------------------------------------------------
// Fused Gram-matrix + top-k KNN.  argmin(dist) == argmax(xi.xj - 0.5|xj|^2).
// Block = 16 rows, 4 waves; each wave computes TWO 16-col tiles per sweep
// (128 cols / 2 barriers).  Scores go to bank-padded LDS; 64 selector threads
// (4 per row, 32 cols each) keep private sorted top-10 lists in registers,
// merged once at the end.  X is L2-resident (<=4MB << 192MB L2) -> kernel is
// compute-bound on v_wmma_f32_16x16x32_f16.
// ---------------------------------------------------------------------------
template<int D>
__global__ __launch_bounds__(128)
void knn_kernel(const _Float16* __restrict__ X, const float* __restrict__ sqn,
                int* __restrict__ nbr) {
  constexpr int KS = D / 32;
  constexpr int SS = 132;                      // score stride: 132 % 64 = 4 -> no bank clash
  __shared__ float scores[16 * SS];
  __shared__ float pv[64][KNNK];
  __shared__ int   pi[64][KNNK];
  const int ib   = blockIdx.x * 16;
  const int tid  = threadIdx.x;
  const int wave = tid >> 5;
  const int lane = tid & 31;
  const int rl   = lane & 15;
  const int hi16 = (lane >> 4) & 1;

  // A fragments for rows ib..ib+15 (K contiguous: 0..15 / 16..31, matches B)
  v16h afrag[KS];
  {
    const _Float16* ap = X + (ib + rl) * D + hi16 * 16;
    #pragma unroll
    for (int s = 0; s < KS; ++s) {
      v8h lo = *(const v8h*)(ap + s * 32);
      v8h hi = *(const v8h*)(ap + s * 32 + 8);
      afrag[s] = make_v16h(lo, hi);
    }
  }

  float bestv[KNNK]; int besti[KNNK];
  #pragma unroll
  for (int q = 0; q < KNNK; ++q) { bestv[q] = -3.0e38f; besti[q] = 0; }

  for (int t = 0; t < NN / 128; ++t) {
    #pragma unroll
    for (int u = 0; u < 2; ++u) {
      const int jb = (t * 8 + wave * 2 + u) * 16;
      v8f acc = {0.f, 0.f, 0.f, 0.f, 0.f, 0.f, 0.f, 0.f};
      // B fragment: col = lane&15, K contiguous (0..15 / 16..31)
      const _Float16* bp = X + (jb + rl) * D + hi16 * 16;
      #pragma unroll
      for (int s = 0; s < KS; ++s) {
        v16h bfrag = *(const v16h*)(bp + s * 32);
        acc = __builtin_amdgcn_wmma_f32_16x16x32_f16(false, afrag[s], false, bfrag,
                                                     (short)0, acc, false, false);
      }
      const int jcol = jb + rl;
      const float halfsq = 0.5f * sqn[jcol];
      #pragma unroll
      for (int r = 0; r < 8; ++r) {
        const int rowt = r + hi16 * 8;
        float sc = acc[r] - halfsq;
        if (jcol == ib + rowt) sc = -3.0e38f;    // exclude self
        scores[rowt * SS + wave * 32 + u * 16 + rl] = sc;
      }
    }
    __syncthreads();
    if (tid < 64) {
      const int row = tid & 15;
      const int c0  = (tid >> 4) * 32;
      for (int c = 0; c < 32; ++c) {
        const float v = scores[row * SS + c0 + c];
        if (v > bestv[KNNK - 1]) {
          const int cc = c0 + c;
          bestv[KNNK - 1] = v;
          besti[KNNK - 1] = (t * 8 + (cc >> 4)) * 16 + (cc & 15);
          #pragma unroll
          for (int q = KNNK - 1; q > 0; --q) {
            if (bestv[q] > bestv[q - 1]) {
              float tv = bestv[q]; bestv[q] = bestv[q - 1]; bestv[q - 1] = tv;
              int   ti = besti[q]; besti[q] = besti[q - 1]; besti[q - 1] = ti;
            }
          }
        }
      }
    }
    __syncthreads();
  }

  // publish partial lists, then merge 4 quarters per row
  if (tid < 64) {
    #pragma unroll
    for (int q = 0; q < KNNK; ++q) { pv[tid][q] = bestv[q]; pi[tid][q] = besti[q]; }
  }
  __syncthreads();
  if (tid < 16) {
    float fv[KNNK]; int fi[KNNK];
    #pragma unroll
    for (int q = 0; q < KNNK; ++q) { fv[q] = -3.0e38f; fi[q] = 0; }
    for (int p = 0; p < 4; ++p) {
      #pragma unroll
      for (int q = 0; q < KNNK; ++q) {
        const float v = pv[tid + p * 16][q];
        const int   j = pi[tid + p * 16][q];
        if (v > fv[KNNK - 1]) {
          fv[KNNK - 1] = v; fi[KNNK - 1] = j;
          #pragma unroll
          for (int s = KNNK - 1; s > 0; --s) {
            if (fv[s] > fv[s - 1]) {
              float tv = fv[s]; fv[s] = fv[s - 1]; fv[s - 1] = tv;
              int   ti = fi[s]; fi[s] = fi[s - 1]; fi[s - 1] = ti;
            }
          }
        }
      }
    }
    #pragma unroll
    for (int q = 0; q < KNNK; ++q) nbr[(ib + tid) * KNNK + q] = fi[q];
  }
}

// ---------------------------------------------------------------------------
// EdgeConv: build e = [x_i, x_j - x_i] for 8 nodes (80 edges) in LDS (f16),
// WMMA the [80 x 2*DIN] @ [2*DIN x 64] MLP (5 row-tiles x 4 col-tiles,
// column-tile per wave), leaky-ReLU epilogue, then max over k=10 edges.
// Optionally emits f16 copy + squared norms for the next KNN layer.
// ---------------------------------------------------------------------------
template<int DIN, bool EMIT>
__global__ __launch_bounds__(128)
void edgeconv_kernel(const float* __restrict__ Xf, const int* __restrict__ nbr,
                     const _Float16* __restrict__ WeT, const float* __restrict__ be,
                     float* __restrict__ Gf, _Float16* __restrict__ Gh,
                     float* __restrict__ Gsq) {
  constexpr int EF  = 2 * DIN;
  constexpr int ES  = EF + 16;                 // padded stride, keeps 32B align
  constexpr int HTS = 68;                      // 68 % 64 = 4 -> bank-safe
  __shared__ _Float16 e[80 * ES];
  __shared__ float ht[80 * HTS];
  __shared__ float sqacc[8];
  const int nb  = blockIdx.x * 8;
  const int tid = threadIdx.x;

  for (int q = tid; q < 80 * EF; q += 128) {
    const int r = q / EF, c = q - r * EF;
    const int nl = r / KNNK;
    const int node = nb + nl;
    float v;
    if (c < DIN) {
      v = Xf[node * DIN + c];
    } else {
      const int j = nbr[node * KNNK + (r - nl * KNNK)];
      v = Xf[j * DIN + (c - DIN)] - Xf[node * DIN + (c - DIN)];
    }
    e[r * ES + c] = (_Float16)v;
  }
  if (tid < 8) sqacc[tid] = 0.f;
  __syncthreads();

  const int wave = tid >> 5, lane = tid & 31;
  const int rl = lane & 15, hi16 = (lane >> 4) & 1;
  #pragma unroll
  for (int j5 = 0; j5 < 5; ++j5) {
    const int jid = wave + 4 * j5;             // 0..19 jobs
    const int rt = jid >> 2, ct = jid & 3;
    v8f acc = {0.f, 0.f, 0.f, 0.f, 0.f, 0.f, 0.f, 0.f};
    const _Float16* ap = &e[(rt * 16 + rl) * ES + hi16 * 16];
    const _Float16* bp = WeT + (ct * 16 + rl) * EF + hi16 * 16;
    #pragma unroll
    for (int s = 0; s < EF / 32; ++s) {
      v8h lo = *(const v8h*)(ap + s * 32);
      v8h hi = *(const v8h*)(ap + s * 32 + 8);
      v16h a = make_v16h(lo, hi);
      v16h b = *(const v16h*)(bp + s * 32);
      acc = __builtin_amdgcn_wmma_f32_16x16x32_f16(false, a, false, b,
                                                   (short)0, acc, false, false);
    }
    const int col = ct * 16 + rl;
    const float bias = be[col];
    #pragma unroll
    for (int r = 0; r < 8; ++r) {
      float v = acc[r] + bias;
      v = v > 0.f ? v : 0.01f * v;             // leaky before max-agg
      ht[(rt * 16 + r + hi16 * 8) * HTS + col] = v;
    }
  }
  __syncthreads();

  for (int q = tid; q < 8 * 64; q += 128) {
    const int nl = q >> 6, col = q & 63;
    float m = ht[(nl * KNNK) * HTS + col];
    #pragma unroll
    for (int s = 1; s < KNNK; ++s) m = fmaxf(m, ht[(nl * KNNK + s) * HTS + col]);
    Gf[(nb + nl) * DGCD + col] = m;
    if (EMIT) {
      Gh[(nb + nl) * DGCD + col] = (_Float16)m;
      atomicAdd(&sqacc[nl], m * m);
    }
  }
  if (EMIT) {
    __syncthreads();
    if (tid < 8) Gsq[nb + tid] = sqacc[tid];
  }
}

// ---------------------------------------------------------------------------
// Classifier: concat(g1,g2) [N,128] @ Wc [128,16] + bc.
// ---------------------------------------------------------------------------
__global__ __launch_bounds__(128)
void classifier_kernel(const float* __restrict__ G1, const float* __restrict__ G2,
                       const float* __restrict__ Wc, const float* __restrict__ bc,
                       float* __restrict__ out) {
  int gid = blockIdx.x * 128 + threadIdx.x;
  int n = gid >> 4, c = gid & 15;
  if (n >= NN) return;
  float v = bc[c];
  for (int k = 0; k < DGCD; ++k) v = fmaf(G1[n * DGCD + k], Wc[k * NCLSD + c], v);
  for (int k = 0; k < DGCD; ++k) v = fmaf(G2[n * DGCD + k], Wc[(DGCD + k) * NCLSD + c], v);
  out[n * NCLSD + c] = v;
}

// ---------------------------------------------------------------------------
extern "C" void kernel_launch(void* const* d_in, const int* in_sizes, int n_in,
                              void* d_out, int out_size, void* d_ws, size_t ws_size,
                              hipStream_t stream) {
  const float* nf  = (const float*)d_in[0];
  const float* rf  = (const float*)d_in[1];
  const float* tf  = (const float*)d_in[2];
  const float* Wb  = (const float*)d_in[3];
  const float* bb  = (const float*)d_in[4];
  const float* Wr  = (const float*)d_in[5];
  const float* br  = (const float*)d_in[6];
  const float* Wt  = (const float*)d_in[7];
  const float* bt  = (const float*)d_in[8];
  const float* Wf  = (const float*)d_in[9];
  const float* bf  = (const float*)d_in[10];
  const float* We1 = (const float*)d_in[11];
  const float* be1 = (const float*)d_in[12];
  const float* We2 = (const float*)d_in[13];
  const float* be2 = (const float*)d_in[14];
  const float* Wc  = (const float*)d_in[15];
  const float* bc  = (const float*)d_in[16];

  char* ws = (char*)d_ws;
  size_t off = 0;
  auto alloc = [&](size_t bytes) -> void* {
    void* p = ws + off;
    off += (bytes + 255) & ~(size_t)255;
    return p;
  };
  float*    Hf   = (float*)   alloc((size_t)NN * HIDD * 4);
  _Float16* Hh   = (_Float16*)alloc((size_t)NN * HIDD * 2);
  float*    Hsq  = (float*)   alloc((size_t)NN * 4);
  float*    G1f  = (float*)   alloc((size_t)NN * DGCD * 4);
  _Float16* G1h  = (_Float16*)alloc((size_t)NN * DGCD * 2);
  float*    G1sq = (float*)   alloc((size_t)NN * 4);
  float*    G2f  = (float*)   alloc((size_t)NN * DGCD * 4);
  int*      idx1 = (int*)     alloc((size_t)NN * KNNK * 4);
  int*      idx2 = (int*)     alloc((size_t)NN * KNNK * 4);
  _Float16* WeT1 = (_Float16*)alloc(64 * 256 * 2);
  _Float16* WeT2 = (_Float16*)alloc(64 * 128 * 2);

  prep_weights_kernel<<<(64 * 256 + 64 * 128 + 255) / 256, 256, 0, stream>>>(
      We1, We2, WeT1, WeT2);
  embed_fuse_kernel<<<2048, 192, 0, stream>>>(
      nf, rf, tf, Wb, bb, Wr, br, Wt, bt, Wf, bf, Hf, Hh, Hsq);
  knn_kernel<HIDD><<<NN / 16, 128, 0, stream>>>(Hh, Hsq, idx1);
  edgeconv_kernel<HIDD, true><<<NN / 8, 128, 0, stream>>>(
      Hf, idx1, WeT1, be1, G1f, G1h, G1sq);
  knn_kernel<DGCD><<<NN / 16, 128, 0, stream>>>(G1h, G1sq, idx2);
  edgeconv_kernel<DGCD, false><<<NN / 8, 128, 0, stream>>>(
      G1f, idx2, WeT2, be2, G2f, (_Float16*)nullptr, (float*)nullptr);
  classifier_kernel<<<(NN * NCLSD) / 128, 128, 0, stream>>>(
      G1f, G2f, Wc, bc, (float*)d_out);
}
